// Sequence_28192165331002
// MI455X (gfx1250) — matrix operand order repaired and checked by reading, not compiled, not run on hardware
//
#include <hip/hip_runtime.h>

typedef __attribute__((ext_vector_type(16))) __bf16 v16bf;
typedef __attribute__((ext_vector_type(8)))  __bf16 v8bf;
typedef __attribute__((ext_vector_type(8)))  float  v8f;
typedef __attribute__((ext_vector_type(4)))  unsigned v4u;
typedef __attribute__((ext_vector_type(8)))  int      v8i;
typedef __attribute__((ext_vector_type(4)))  int      v4i;

#define T_DIM 512
#define B_DIM 256
#define H_DIM 256
#define NWG   32   // 8 batch tiles x 4 h-col tiles

// ---------------------------------------------------------------------------
// Fast activations on CDNA5 hardware transcendentals (TRANS32 ops co-issue
// with WMMA). Guarded so missing builtins degrade to libm.
// ---------------------------------------------------------------------------
__device__ __forceinline__ float fast_exp2(float x) {
#if __has_builtin(__builtin_amdgcn_exp2f)
  return __builtin_amdgcn_exp2f(x);
#else
  return __expf(x * 0.6931471805599453f);
#endif
}
__device__ __forceinline__ float fast_rcp(float x) {
#if __has_builtin(__builtin_amdgcn_rcpf)
  return __builtin_amdgcn_rcpf(x);
#else
  return 1.f / x;
#endif
}
__device__ __forceinline__ float sigf(float x) {
  return fast_rcp(1.f + fast_exp2(x * -1.44269504088896340736f));
}
__device__ __forceinline__ float tanh_fast(float x) {
#if __has_builtin(__builtin_amdgcn_tanhf)
  return __builtin_amdgcn_tanhf(x);
#else
  // tanh(x) = 1 - 2/(exp2(x * 2*log2(e)) + 1)
  float e = fast_exp2(x * 2.88539008177792681472f);
  return 1.f - 2.f * fast_rcp(e + 1.f);
#endif
}

// ---------------------------------------------------------------------------
// Prep: pack [w_ih | pad | w_hh] into one bf16 matrix (4H x Kpad), sum biases.
// ---------------------------------------------------------------------------
__global__ void pack_weights_kernel(const float* __restrict__ wih,
                                    const float* __restrict__ whh,
                                    const float* __restrict__ bih,
                                    const float* __restrict__ bhh,
                                    __bf16* __restrict__ wcat,
                                    float* __restrict__ bias,
                                    int in_dim, int Ipad) {
  const int Kpad = Ipad + H_DIM;
  const int total = 4 * H_DIM * Kpad;
  for (int idx = blockIdx.x * blockDim.x + threadIdx.x; idx < total;
       idx += gridDim.x * blockDim.x) {
    int g = idx / Kpad;
    int k = idx - g * Kpad;
    float v = 0.f;
    if (k < in_dim)       v = wih[g * in_dim + k];
    else if (k >= Ipad)   v = whh[g * H_DIM + (k - Ipad)];
    wcat[idx] = (__bf16)v;
    if (idx < 4 * H_DIM) bias[idx] = bih[idx] + bhh[idx];
  }
}

// x: (T,B,12) fp32 -> xp: (T,B,32) bf16 zero-padded
__global__ void pack_input_kernel(const float* __restrict__ x,
                                  __bf16* __restrict__ xp) {
  const int total = T_DIM * B_DIM * 32;
  for (int idx = blockIdx.x * blockDim.x + threadIdx.x; idx < total;
       idx += gridDim.x * blockDim.x) {
    int k  = idx & 31;
    int tb = idx >> 5;
    xp[idx] = (__bf16)((k < 12) ? x[tb * 12 + k] : 0.f);
  }
}

__global__ void init_state_kernel(__bf16* __restrict__ h, float* __restrict__ c,
                                  unsigned* __restrict__ bar) {
  int idx = blockIdx.x * blockDim.x + threadIdx.x;
  if (idx < B_DIM * H_DIM) { h[idx] = (__bf16)0.f; c[idx] = 0.f; }
  if (idx == 0) { bar[0] = 0u; bar[1] = 0u; }
}

// ---------------------------------------------------------------------------
// Persistent recurrent kernel: 32 WGs x 128 threads, weights resident in LDS
// (loaded via the Tensor Data Mover when available), one grid-wide barrier
// per timestep.  WG tile: 32 batch rows x 64 h-cols (x4 gates); wave: 16x32.
// ---------------------------------------------------------------------------
__global__ void __launch_bounds__(128)
lstm_layer_kernel(const __bf16* __restrict__ X, int Ipad,
                  const __bf16* __restrict__ Wcat,
                  const float* __restrict__ bias,
                  __bf16* __restrict__ h, float* __restrict__ c,
                  __bf16* __restrict__ Y,
                  unsigned* __restrict__ bar, int nwg) {
  extern __shared__ char smem[];
  const int Kpad = Ipad + H_DIM;
  const int tid  = threadIdx.x;
  const int lane = tid & 31;
  const int w    = tid >> 5;
  const int bx   = blockIdx.x & 7;     // batch tile
  const int nx   = blockIdx.x >> 3;    // h-col tile
  const int mrow = bx * 32 + (w & 1) * 16;
  const int wn   = w >> 1;
  const int ncol = nx * 64 + wn * 32;  // wave's base h-col
  const int l15  = lane & 15;
  const int s16  = lane >> 4;          // K sub-block selector

  // ---- preload this WG's weight slice (4 gates x 64 rows x Kpad bf16) to LDS
#if __has_builtin(__builtin_amdgcn_tensor_load_to_lds) && \
    __has_builtin(__builtin_amdgcn_s_wait_tensorcnt)
  if (w == 0) {
    // One 2D TDM descriptor per gate block: tile = Kpad x 64 rows, 2B elems.
    const unsigned lds0 = (unsigned)(unsigned long long)(uintptr_t)smem;
    const unsigned long long gbase =
        (unsigned long long)(uintptr_t)Wcat +
        (unsigned long long)(nx * 64) * (unsigned)Kpad * 2ull;
    const unsigned rowBytes = (unsigned)Kpad * 2u;
#pragma unroll 1
    for (int g = 0; g < 4; ++g) {
      unsigned long long gaddr =
          gbase + (unsigned long long)g * H_DIM * (unsigned)Kpad * 2ull;
      v4u g0;
      g0[0] = 1u;                                   // count=1, user mode
      g0[1] = lds0 + (unsigned)g * 64u * rowBytes;  // lds_addr (bytes)
      g0[2] = (unsigned)(gaddr & 0xFFFFFFFFull);    // global_addr[31:0]
      g0[3] = (unsigned)((gaddr >> 32) & 0x1FFFFFFull) | (2u << 30); // type=2
      v8i g1;
      g1[0] = (int)(1u << 16);                      // data_size = 2 bytes
      g1[1] = (int)(((unsigned)Kpad & 0xFFFFu) << 16);          // tensor_dim0 lo
      g1[2] = (int)(((unsigned)Kpad >> 16) | (1024u << 16));    // dim0 hi | tensor_dim1 lo
      g1[3] = (int)(((unsigned)Kpad & 0xFFFFu) << 16);          // dim1 hi=0 | tile_dim0
      g1[4] = 64;                                   // tile_dim1=64, tile_dim2=0
      g1[5] = (int)(unsigned)Kpad;                  // tensor_dim0_stride lo
      g1[6] = 0;
      g1[7] = 0;
      v4i zz4 = {0, 0, 0, 0};
      v8i zz8 = {0, 0, 0, 0, 0, 0, 0, 0};
      __builtin_amdgcn_tensor_load_to_lds(g0, g1, zz4, zz4, zz8, 0);
    }
    __builtin_amdgcn_s_wait_tensorcnt(0);
  }
#else
  {
    const int chunksPerRow = Kpad / 8;        // 16-byte chunks per row
    const int nchunks = 256 * chunksPerRow;
    for (int i = tid; i < nchunks; i += 128) {
      int rl = i / chunksPerRow;              // local row 0..255
      int ck = i - rl * chunksPerRow;
      int gate = rl >> 6;
      int grow = gate * H_DIM + nx * 64 + (rl & 63);
      const v8bf* src = (const v8bf*)(Wcat + (size_t)grow * Kpad + ck * 8);
      *(v8bf*)(smem + (size_t)rl * Kpad * 2 + (size_t)ck * 16) = *src;
    }
  }
#endif
  __syncthreads();

  // per-lane biases: gate g, n-tile nt -> gate column ncol + nt*16 + (lane&15)
  float bv[4][2];
#pragma unroll
  for (int g = 0; g < 4; ++g)
#pragma unroll
    for (int nt = 0; nt < 2; ++nt)
      bv[g][nt] = bias[g * H_DIM + ncol + nt * 16 + l15];

  const int am = mrow + l15;       // A-matrix (batch) row this lane loads
  const int nK = Kpad / 32;
  const int nXtiles = Ipad / 32;
  const v8f vzero = {0.f, 0.f, 0.f, 0.f, 0.f, 0.f, 0.f, 0.f};

  for (int t = 0; t < T_DIM; ++t) {
    v8f acc[4][2];
#pragma unroll
    for (int g = 0; g < 4; ++g) { acc[g][0] = vzero; acc[g][1] = vzero; }

    const __bf16* xrow = X + ((size_t)t * B_DIM + am) * Ipad;

#pragma unroll 2
    for (int kt = 0; kt < nK; ++kt) {
      const int kbase = kt * 32;
      // A tile 16x32: lanes m / m+16 hold K {0..15} / {16..31}
      const __bf16* arow = (kt < nXtiles)
                             ? (xrow + kbase)
                             : (h + (size_t)am * H_DIM + (kbase - Ipad));
      union { v16bf v; v8bf p[2]; } A;
      A.p[0] = *(const v8bf*)(arow + s16 * 16);
      A.p[1] = *(const v8bf*)(arow + s16 * 16 + 8);

#pragma unroll
      for (int g = 0; g < 4; ++g) {
#pragma unroll
        for (int nt = 0; nt < 2; ++nt) {
          // B tile 32x16 from LDS: lane = out channel, lanes 16-31 take K+16
          const int rl = wn * 32 + nt * 16 + l15;
          const v16bf* bp = (const v16bf*)(smem +
              ((size_t)(g * 64 + rl) * Kpad + kbase + s16 * 16) * 2);
          v16bf Bm = *bp;
          acc[g][nt] = __builtin_amdgcn_wmma_f32_16x16x32_bf16(
              false, A.v, false, Bm, (short)0, acc[g][nt], false, false);
        }
      }
    }

    // prefetch next timestep's x rows while we do the cell update
    if (t + 1 < T_DIM)
      __builtin_prefetch(X + ((size_t)(t + 1) * B_DIM + am) * Ipad, 0, 1);

    // ---- elementwise LSTM cell: D layout -> rows j (lanes 0-15) / j+8 (16-31)
#pragma unroll
    for (int nt = 0; nt < 2; ++nt) {
      const int n = ncol + nt * 16 + l15;
#pragma unroll
      for (int j = 0; j < 8; ++j) {
        const int m   = mrow + j + s16 * 8;
        const int idx = m * H_DIM + n;
        float iv = sigf(acc[0][nt][j] + bv[0][nt]);
        float fv = sigf(acc[1][nt][j] + bv[1][nt]);
        float gv = tanh_fast(acc[2][nt][j] + bv[2][nt]);
        float ov = sigf(acc[3][nt][j] + bv[3][nt]);
        float cn = fv * c[idx] + iv * gv;
        c[idx] = cn;
        float hn = ov * tanh_fast(cn);
        __bf16 hb = (__bf16)hn;
        h[idx] = hb;
        Y[(size_t)t * (B_DIM * H_DIM) + idx] = hb;
      }
    }

    // ---- grid-wide barrier (generation-count sense reversal)
    __threadfence();
    __syncthreads();
    if (tid == 0) {
      unsigned g0 = atomicAdd(&bar[1], 0u);
      if (atomicAdd(&bar[0], 1u) == (unsigned)nwg - 1u) {
        atomicExch(&bar[0], 0u);
        atomicAdd(&bar[1], 1u);
      } else {
        while (atomicAdd(&bar[1], 0u) == g0) __builtin_amdgcn_s_sleep(1);
      }
    }
    __syncthreads();
  }
}

// ---------------------------------------------------------------------------
// Final linear: out[t,b] = Y[t,b,:] . w_lin + b_lin   (one wave per output)
// ---------------------------------------------------------------------------
__global__ void linear_kernel(const __bf16* __restrict__ Y,
                              const float* __restrict__ wlin,
                              const float* __restrict__ blin,
                              float* __restrict__ out) {
  int wave = blockIdx.x * (blockDim.x >> 5) + (threadIdx.x >> 5);
  int lane = threadIdx.x & 31;
  if (wave >= T_DIM * B_DIM) return;
  const __bf16* y = Y + (size_t)wave * H_DIM;
  float s = 0.f;
  for (int n = lane; n < H_DIM; n += 32) s += (float)y[n] * wlin[n];
#pragma unroll
  for (int off = 16; off > 0; off >>= 1) s += __shfl_xor(s, off, 32);
  if (lane == 0) out[wave] = s + blin[0];
}

// ---------------------------------------------------------------------------
extern "C" void kernel_launch(void* const* d_in, const int* in_sizes, int n_in,
                              void* d_out, int out_size, void* d_ws, size_t ws_size,
                              hipStream_t stream) {
  const float* input  = (const float*)d_in[0];
  const float* wih[3] = {(const float*)d_in[1], (const float*)d_in[5], (const float*)d_in[9]};
  const float* whh[3] = {(const float*)d_in[2], (const float*)d_in[6], (const float*)d_in[10]};
  const float* bih[3] = {(const float*)d_in[3], (const float*)d_in[7], (const float*)d_in[11]};
  const float* bhh[3] = {(const float*)d_in[4], (const float*)d_in[8], (const float*)d_in[12]};
  const float* wlin   = (const float*)d_in[13];
  const float* blin   = (const float*)d_in[14];
  float* out = (float*)d_out;

  char* ws = (char*)d_ws;
  size_t off = 0;
  auto alloc = [&](size_t bytes) -> char* {
    off = (off + 255) & ~(size_t)255;
    char* p = ws + off;
    off += bytes;
    return p;
  };

  unsigned* bar = (unsigned*)alloc(256);
  int Ipad[3] = {32, 256, 256};
  __bf16* wcat[3];
  float*  bias[3];
  for (int l = 0; l < 3; ++l) {
    int Kpad = Ipad[l] + H_DIM;
    wcat[l] = (__bf16*)alloc((size_t)4 * H_DIM * Kpad * sizeof(__bf16));
    bias[l] = (float*)alloc((size_t)4 * H_DIM * sizeof(float));
  }
  __bf16* hbuf = (__bf16*)alloc((size_t)B_DIM * H_DIM * sizeof(__bf16));
  float*  cbuf = (float*)alloc((size_t)B_DIM * H_DIM * sizeof(float));
  __bf16* x0p  = (__bf16*)alloc((size_t)T_DIM * B_DIM * 32 * sizeof(__bf16));
  __bf16* bufA = (__bf16*)alloc((size_t)T_DIM * B_DIM * H_DIM * sizeof(__bf16));
  __bf16* bufB = (__bf16*)alloc((size_t)T_DIM * B_DIM * H_DIM * sizeof(__bf16));

  pack_input_kernel<<<1024, 256, 0, stream>>>(input, x0p);
  for (int l = 0; l < 3; ++l)
    pack_weights_kernel<<<1024, 256, 0, stream>>>(wih[l], whh[l], bih[l], bhh[l],
                                                  wcat[l], bias[l],
                                                  l == 0 ? 12 : H_DIM, Ipad[l]);

  const __bf16* Xl[3] = {x0p, bufA, bufB};
  __bf16*       Yl[3] = {bufA, bufB, bufA};
  for (int l = 0; l < 3; ++l) {
    init_state_kernel<<<(B_DIM * H_DIM + 255) / 256, 256, 0, stream>>>(hbuf, cbuf, bar);
    size_t lds = (size_t)256 * (Ipad[l] + H_DIM) * sizeof(__bf16);  // 144KB / 256KB
    lstm_layer_kernel<<<NWG, 128, lds, stream>>>(Xl[l], Ipad[l], wcat[l], bias[l],
                                                 hbuf, cbuf, Yl[l], bar, NWG);
  }

  // 131072 outputs, 8 waves per 256-thread block
  linear_kernel<<<(T_DIM * B_DIM) / 8, 256, 0, stream>>>(bufA, wlin, blin, out);
}
